// SelectiveSSM_76493367541788
// MI455X (gfx1250) — matrix operand order repaired and checked by reading, not compiled, not run on hardware
//
#include <hip/hip_runtime.h>
#include <cstdint>
#include <cstddef>

// ---------------------------------------------------------------------------
// Selective SSM (Mamba block) forward for MI455X (gfx1250), wave32 + WMMA.
// GEMMs: v_wmma_f32_16x16x32_bf16, double-buffered LDS fed by
// GLOBAL_LOAD_ASYNC_TO_LDS_B128 (ASYNCcnt-tracked direct L2->LDS DMA).
// Scan: 3-phase chunked associative scan.
// ---------------------------------------------------------------------------

typedef __bf16 bf16_t;
typedef __attribute__((ext_vector_type(16))) __bf16 v16bf;
typedef __attribute__((ext_vector_type(8)))  float  v8f;
typedef __attribute__((ext_vector_type(4)))  unsigned int v4u;

// Problem constants (match reference)
constexpr int L_SEQ   = 2048;
constexpr int D_MODEL = 1024;
constexpr int D_INNER = 2048;
constexpr int N_ST    = 16;
constexpr int K_CONV  = 4;
constexpr int N_CHUNK = 16;               // scan chunks
constexpr int CHUNK_L = L_SEQ / N_CHUNK;  // 128

// Workspace layout (bytes)
constexpr size_t MBy        = 1ull << 20;
constexpr size_t OFF_XBF    = 0;            // 4 MB  (x in bf16; dead after GEMM1)
constexpr size_t OFF_YBF    = 0;            // 8 MB  (reuses XBF region after GEMM1)
constexpr size_t OFF_WINT   = 8  * MBy;     // 8 MB  (W_in^T  bf16, 4096x1024)
constexpr size_t OFF_WOUTT  = 16 * MBy;     // 4 MB  (W_out^T bf16, 1024x2048)
constexpr size_t OFF_XZ     = 20 * MBy;     // 32 MB (xz f32, 2048x4096)
constexpr size_t OFF_XO     = 52 * MBy;     // 16 MB (x_ post conv+silu, f32)
constexpr size_t OFF_BCD    = 68 * MBy;     // 264 KB (bcd f32, 2048x33)
constexpr size_t OFF_APROD  = 69 * MBy;     // 2 MB  (chunk A-products)
constexpr size_t OFF_HEND   = 71 * MBy;     // 2 MB  (chunk end states)
constexpr size_t OFF_HINIT  = 73 * MBy;     // 2 MB  (chunk init states)
constexpr size_t OFF_YRAW   = 75 * MBy;     // 16 MB (y[l,d] pre-reduction)
constexpr size_t OFF_YS     = 91 * MBy;     // 8 KB  (per-l scalar)

__device__ __forceinline__ float silu_f(float v) {
    return v / (1.0f + __expf(-v));
}
__device__ __forceinline__ float softplus_f(float v) {
    return (v > 20.0f) ? v : log1pf(__expf(v));
}

// ---------------------------------------------------------------------------
// Conversion / transpose kernels
// ---------------------------------------------------------------------------
__global__ __launch_bounds__(256) void k_cvt_bf16(const float* __restrict__ in,
                                                  bf16_t* __restrict__ out, int n) {
    int i = blockIdx.x * 256 + threadIdx.x;
    if (i < n) out[i] = (bf16_t)in[i];
}

// in: R x C f32 row-major  ->  out: C x R bf16 row-major (i.e. in^T)
__global__ __launch_bounds__(256) void k_transpose_cvt(const float* __restrict__ in,
                                                       bf16_t* __restrict__ out,
                                                       int R, int C) {
    int i = blockIdx.x * 256 + threadIdx.x;
    if (i < R * C) {
        int r = i / C, c = i % C;
        out[(size_t)c * R + r] = (bf16_t)in[i];
    }
}

// ---------------------------------------------------------------------------
// WMMA bf16 GEMM:  C[M,N] = A[M,K] * B[K,N],  B given pre-transposed (N x K).
// Block 256 threads = 8 wave32; tile 128x128, BK=32; wave = 64x32 (4x2 frags).
// Double-buffered LDS filled by GLOBAL_LOAD_ASYNC_TO_LDS_B128:
//   issue tile t+1 -> buf^1 while computing tile t from buf.
//   s_wait_asynccnt 4 (async loads complete in order) waits only the old tile.
// Dynamic LDS so the async op's per-lane LDS byte offsets are known (base 0).
// ---------------------------------------------------------------------------
#define BM 128
#define BN 128
#define BK 32
#define LDSS 40                              // BK + 8 bf16 pad -> 80 B row stride
constexpr int      TILE_ELEMS = BM * LDSS;   // 5120 bf16 per tile
constexpr unsigned TILE_BYTES = TILE_ELEMS * 2;      // 10240
constexpr unsigned BUF_BYTES  = 2 * TILE_BYTES;      // 20480 (A tile + B tile)
constexpr unsigned GEMM_LDS   = 2 * BUF_BYTES;       // 40960 total

__global__ __launch_bounds__(256)
void k_gemm_bf16(const bf16_t* __restrict__ A,    // M x K
                 const bf16_t* __restrict__ Bt,   // N x K
                 float* __restrict__ C,           // M x N
                 int M, int N, int K) {
    extern __shared__ bf16_t smem[];  // [buf0 A][buf0 B][buf1 A][buf1 B]

    const int tid  = threadIdx.x;
    const int lane = tid & 31;
    const int wave = tid >> 5;
    const int wm   = wave >> 2;   // 0..1
    const int wn   = wave & 3;    // 0..3
    const int hilo = lane >> 4;   // 0/1 half-wave
    const int l16  = lane & 15;

    const int bm = blockIdx.y * BM;
    const int bn = blockIdx.x * BN;

    v8f acc[4][2];
    const v8f vzero = {0.f, 0.f, 0.f, 0.f, 0.f, 0.f, 0.f, 0.f};
#pragma unroll
    for (int i = 0; i < 4; ++i)
#pragma unroll
        for (int j = 0; j < 2; ++j) acc[i][j] = vzero;

    // Tile staging: 128 rows x 32 bf16 = 512 x 16B chunks; 2 chunks/thread/tile.
    const int r0 = tid >> 2,         c0 = tid & 3;
    const int r1 = (tid + 256) >> 2, c1 = (tid + 256) & 3;

    // per-thread global source pointers (chunk base; advances by BK per tile)
    const bf16_t* gA0 = &A [(size_t)(bm + r0) * K + c0 * 8];
    const bf16_t* gA1 = &A [(size_t)(bm + r1) * K + c1 * 8];
    const bf16_t* gB0 = &Bt[(size_t)(bn + r0) * K + c0 * 8];
    const bf16_t* gB1 = &Bt[(size_t)(bn + r1) * K + c1 * 8];

    // per-thread LDS byte offsets within one buffer
    const unsigned lA0 = (unsigned)(r0 * (LDSS * 2) + c0 * 16);
    const unsigned lA1 = (unsigned)(r1 * (LDSS * 2) + c1 * 16);
    const unsigned lB0 = TILE_BYTES + lA0;
    const unsigned lB1 = TILE_BYTES + lA1;

#define ISSUE_TILE(k0, bufbase)                                                 \
    do {                                                                        \
        asm volatile("global_load_async_to_lds_b128 %0, %1, off"                \
                     :: "v"((bufbase) + lA0), "v"(gA0 + (k0)) : "memory");      \
        asm volatile("global_load_async_to_lds_b128 %0, %1, off"                \
                     :: "v"((bufbase) + lA1), "v"(gA1 + (k0)) : "memory");      \
        asm volatile("global_load_async_to_lds_b128 %0, %1, off"                \
                     :: "v"((bufbase) + lB0), "v"(gB0 + (k0)) : "memory");      \
        asm volatile("global_load_async_to_lds_b128 %0, %1, off"                \
                     :: "v"((bufbase) + lB1), "v"(gB1 + (k0)) : "memory");      \
    } while (0)

    union Frag { v16bf v; v4u q[2]; };

    const int T = K / BK;
    ISSUE_TILE(0, 0u);

    for (int t = 0; t < T; ++t) {
        const unsigned buf = (unsigned)(t & 1);
        if (t + 1 < T) {
            ISSUE_TILE((t + 1) * BK, (buf ^ 1u) * BUF_BYTES);
            // 4 newer asyncs outstanding; in-order completion => cnt<=4 means
            // the 4 loads of the *current* tile have landed in LDS.
            asm volatile("s_wait_asynccnt 0x4" ::: "memory");
        } else {
            asm volatile("s_wait_asynccnt 0x0" ::: "memory");
        }
        __syncthreads();   // current tile visible to all waves

        const bf16_t* As = &smem[buf * (2 * TILE_ELEMS)];
        const bf16_t* Bs = As + TILE_ELEMS;

        // B frags: lane holds col n=l16, K = hilo*16 + 0..15 (32B contiguous)
        Frag bf[2];
        const int kb = hilo * 16;
#pragma unroll
        for (int fn = 0; fn < 2; ++fn) {
            const bf16_t* p = &Bs[(wn * 32 + fn * 16 + l16) * LDSS + kb];
            bf[fn].q[0] = *(const v4u*)p;
            bf[fn].q[1] = *(const v4u*)(p + 8);
        }
        // A frags: lane holds row m=l16, K = hilo*16 + 0..15 (32B contiguous)
        const int ka = hilo * 16;
#pragma unroll
        for (int fm = 0; fm < 4; ++fm) {
            Frag af;
            const bf16_t* p = &As[(wm * 64 + fm * 16 + l16) * LDSS + ka];
            af.q[0] = *(const v4u*)p;
            af.q[1] = *(const v4u*)(p + 8);
#pragma unroll
            for (int fn = 0; fn < 2; ++fn) {
                acc[fm][fn] = __builtin_amdgcn_wmma_f32_16x16x32_bf16(
                    false, af.v, false, bf[fn].v, (short)0, acc[fm][fn],
                    false, false);
            }
        }
        __syncthreads();   // all waves done with buf before it is re-filled
    }
#undef ISSUE_TILE

    // C/D layout: VGPR r -> row (hilo*8 + r), col l16
    const int mrow = bm + wm * 64 + hilo * 8;
    const int ncol = bn + wn * 32 + l16;
#pragma unroll
    for (int fm = 0; fm < 4; ++fm)
#pragma unroll
        for (int fn = 0; fn < 2; ++fn)
#pragma unroll
            for (int r = 0; r < 8; ++r)
                C[(size_t)(mrow + fm * 16 + r) * N + ncol + fn * 16] = acc[fm][fn][r];
}

// ---------------------------------------------------------------------------
// Causal depthwise conv (K=4) + SiLU.  x_pre = xz[:, 0:2048]
// ---------------------------------------------------------------------------
__global__ __launch_bounds__(256)
void k_conv_silu(const float* __restrict__ xz, const float* __restrict__ cw,
                 const float* __restrict__ cb, float* __restrict__ xo) {
    int i = blockIdx.x * 256 + threadIdx.x;      // i < L*D_INNER
    int l = i >> 11;          // /2048
    int d = i & 2047;
    float acc = cb[d];
#pragma unroll
    for (int k = 0; k < K_CONV; ++k) {
        int ls = l + k - (K_CONV - 1);
        float v = (ls >= 0) ? xz[(size_t)ls * (2 * D_INNER) + d] : 0.0f;
        acc += cw[d * K_CONV + k] * v;
    }
    xo[i] = silu_f(acc);
}

// ---------------------------------------------------------------------------
// bcd = x_ @ W_x  (skinny GEMM, N=33). One block per row l, row staged in LDS.
// ---------------------------------------------------------------------------
__global__ __launch_bounds__(64)
void k_bcd(const float* __restrict__ xo, const float* __restrict__ Wx,
           float* __restrict__ bcd) {
    __shared__ float row[D_INNER];
    const int l = blockIdx.x;
    for (int k = threadIdx.x; k < D_INNER; k += 64)
        row[k] = xo[(size_t)l * D_INNER + k];
    __syncthreads();
    if (threadIdx.x < 33) {
        const int j = threadIdx.x;
        float s = 0.0f;
        for (int k = 0; k < D_INNER; ++k)
            s += row[k] * Wx[k * 33 + j];
        bcd[l * 33 + j] = s;
    }
}

// ---------------------------------------------------------------------------
// Chunked selective scan. Thread t -> (chunk c, channel d, state n):
//   t = c*32768 + d*16 + n  (so aprod/hend/hinit index == t)
// ---------------------------------------------------------------------------
__global__ __launch_bounds__(256)
void k_scan_phase1(const float* __restrict__ bcd, const float* __restrict__ xo,
                   const float* __restrict__ Wdt, const float* __restrict__ bdt,
                   const float* __restrict__ Alog,
                   float* __restrict__ aprod, float* __restrict__ hend) {
    const int t = blockIdx.x * 256 + threadIdx.x;
    const int n = t & 15;
    const int d = (t >> 4) & (D_INNER - 1);
    const int c = t >> 15;
    const float A   = -__expf(Alog[d * N_ST + n]);
    const float wdt = Wdt[d], bd = bdt[d];
    float ap = 1.0f, h = 0.0f;
    const int l0 = c * CHUNK_L;
    for (int l = l0; l < l0 + CHUNK_L; ++l) {
        float dt = softplus_f(bcd[l * 33 + 32] * wdt + bd);
        float ab = __expf(dt * A);
        float bu = dt * bcd[l * 33 + n] * xo[(size_t)l * D_INNER + d];
        ap *= ab;
        h = ab * h + bu;
    }
    aprod[t] = ap;
    hend[t]  = h;
}

__global__ __launch_bounds__(256)
void k_scan_phase2(const float* __restrict__ aprod, const float* __restrict__ hend,
                   float* __restrict__ hinit) {
    const int u = blockIdx.x * 256 + threadIdx.x;   // u < 32768 = (d,n)
    float H = 0.0f;
    hinit[u] = 0.0f;
    for (int c = 1; c < N_CHUNK; ++c) {
        const int p = (c - 1) * (D_INNER * N_ST) + u;
        H = aprod[p] * H + hend[p];
        hinit[c * (D_INNER * N_ST) + u] = H;
    }
}

__global__ __launch_bounds__(256)
void k_scan_phase3(const float* __restrict__ bcd, const float* __restrict__ xo,
                   const float* __restrict__ Wdt, const float* __restrict__ bdt,
                   const float* __restrict__ Alog, const float* __restrict__ hinit,
                   float* __restrict__ yraw) {
    const int t = blockIdx.x * 256 + threadIdx.x;
    const int n = t & 15;                    // == lane & 15
    const int d = (t >> 4) & (D_INNER - 1);
    const int c = t >> 15;
    const float A   = -__expf(Alog[d * N_ST + n]);
    const float wdt = Wdt[d], bd = bdt[d];
    float h = hinit[t];
    const int l0 = c * CHUNK_L;
    for (int l = l0; l < l0 + CHUNK_L; ++l) {
        float dt = softplus_f(bcd[l * 33 + 32] * wdt + bd);
        float ab = __expf(dt * A);
        float bu = dt * bcd[l * 33 + n] * xo[(size_t)l * D_INNER + d];
        h = ab * h + bu;
        // reduce 16 states (lanes within half-wave)
        float s = h;
        s += __shfl_xor(s, 1, 32);
        s += __shfl_xor(s, 2, 32);
        s += __shfl_xor(s, 4, 32);
        s += __shfl_xor(s, 8, 32);
        if (n == 0) yraw[(size_t)l * D_INNER + d] = s;
    }
}

// ys[l] = Csum[l] * mean_d yraw[l,d]
__global__ __launch_bounds__(256)
void k_reduce(const float* __restrict__ yraw, const float* __restrict__ bcd,
              float* __restrict__ ys) {
    __shared__ float red[256];
    const int l = blockIdx.x;
    float s = 0.0f;
    for (int d = threadIdx.x; d < D_INNER; d += 256)
        s += yraw[(size_t)l * D_INNER + d];
    red[threadIdx.x] = s;
    __syncthreads();
    for (int st = 128; st > 0; st >>= 1) {
        if (threadIdx.x < st) red[threadIdx.x] += red[threadIdx.x + st];
        __syncthreads();
    }
    if (threadIdx.x == 0) {
        float cs = 0.0f;
#pragma unroll
        for (int n = 0; n < N_ST; ++n) cs += bcd[l * 33 + 16 + n];
        ys[l] = red[0] * cs * (1.0f / (float)D_INNER);
    }
}

// y = (ys + D*x_) * silu(z), output bf16 for the final WMMA GEMM
__global__ __launch_bounds__(256)
void k_gate(const float* __restrict__ ys, const float* __restrict__ xo,
            const float* __restrict__ Dp, const float* __restrict__ xz,
            bf16_t* __restrict__ ybf) {
    int i = blockIdx.x * 256 + threadIdx.x;   // i < L*D_INNER
    int l = i >> 11;
    int d = i & 2047;
    float z = xz[(size_t)l * (2 * D_INNER) + D_INNER + d];
    float v = (ys[l] + Dp[d] * xo[i]) * silu_f(z);
    ybf[i] = (bf16_t)v;
}

// ---------------------------------------------------------------------------
extern "C" void kernel_launch(void* const* d_in, const int* in_sizes, int n_in,
                              void* d_out, int out_size, void* d_ws, size_t ws_size,
                              hipStream_t stream) {
    (void)in_sizes; (void)n_in; (void)out_size; (void)ws_size;
    const float* x    = (const float*)d_in[0];
    const float* Win  = (const float*)d_in[1];
    const float* cw   = (const float*)d_in[2];
    const float* cb   = (const float*)d_in[3];
    const float* Wx   = (const float*)d_in[4];
    const float* Wdt  = (const float*)d_in[5];
    const float* bdt  = (const float*)d_in[6];
    const float* Alog = (const float*)d_in[7];
    const float* Dp   = (const float*)d_in[8];
    const float* Wout = (const float*)d_in[9];
    float* out = (float*)d_out;
    char*  ws  = (char*)d_ws;

    bf16_t* xbf   = (bf16_t*)(ws + OFF_XBF);
    bf16_t* ybf   = (bf16_t*)(ws + OFF_YBF);
    bf16_t* WinT  = (bf16_t*)(ws + OFF_WINT);
    bf16_t* WoutT = (bf16_t*)(ws + OFF_WOUTT);
    float*  xz    = (float*)(ws + OFF_XZ);
    float*  xo    = (float*)(ws + OFF_XO);
    float*  bcd   = (float*)(ws + OFF_BCD);
    float*  aprod = (float*)(ws + OFF_APROD);
    float*  hend  = (float*)(ws + OFF_HEND);
    float*  hinit = (float*)(ws + OFF_HINIT);
    float*  yraw  = (float*)(ws + OFF_YRAW);
    float*  ys    = (float*)(ws + OFF_YS);

    // 1. precision conversion / weight transposes
    k_cvt_bf16<<<(L_SEQ * D_MODEL) / 256, 256, 0, stream>>>(x, xbf, L_SEQ * D_MODEL);
    k_transpose_cvt<<<(D_MODEL * 2 * D_INNER) / 256, 256, 0, stream>>>(Win, WinT, D_MODEL, 2 * D_INNER);
    k_transpose_cvt<<<(D_INNER * D_MODEL) / 256, 256, 0, stream>>>(Wout, WoutT, D_INNER, D_MODEL);

    // 2. xz = x @ W_in   (2048x1024 * 1024x4096)
    k_gemm_bf16<<<dim3(2 * D_INNER / BN, L_SEQ / BM), 256, GEMM_LDS, stream>>>(
        xbf, WinT, xz, L_SEQ, 2 * D_INNER, D_MODEL);

    // 3. depthwise causal conv + SiLU
    k_conv_silu<<<(L_SEQ * D_INNER) / 256, 256, 0, stream>>>(xz, cw, cb, xo);

    // 4. bcd = x_ @ W_x
    k_bcd<<<L_SEQ, 64, 0, stream>>>(xo, Wx, bcd);

    // 5. chunked selective scan
    k_scan_phase1<<<(N_CHUNK * D_INNER * N_ST) / 256, 256, 0, stream>>>(
        bcd, xo, Wdt, bdt, Alog, aprod, hend);
    k_scan_phase2<<<(D_INNER * N_ST) / 256, 256, 0, stream>>>(aprod, hend, hinit);
    k_scan_phase3<<<(N_CHUNK * D_INNER * N_ST) / 256, 256, 0, stream>>>(
        bcd, xo, Wdt, bdt, Alog, hinit, yraw);

    // 6. per-timestep reduction + C gating
    k_reduce<<<L_SEQ, 256, 0, stream>>>(yraw, bcd, ys);

    // 7. skip + SiLU(z) gate, cast to bf16
    k_gate<<<(L_SEQ * D_INNER) / 256, 256, 0, stream>>>(ys, xo, Dp, xz, ybf);

    // 8. out = y @ W_out  (2048x2048 * 2048x1024)
    k_gemm_bf16<<<dim3(D_MODEL / BN, L_SEQ / BM), 256, GEMM_LDS, stream>>>(
        ybf, WoutT, out, L_SEQ, D_MODEL, D_INNER);
}
